// LGNNPlusRATLayer_53223234732416
// MI455X (gfx1250) — matrix operand order repaired and checked by reading, not compiled, not run on hardware
//
#include <hip/hip_runtime.h>

// ---------------------------------------------------------------------------
// LGNN+RAT layer for MI455X (gfx1250, wave32, WMMA).
// GEMMs: v_wmma_f32_16x16x32_bf16, f32 accumulate. Both GEMM operands are
// bf16 in global memory (weights pre-transposed [N][K]; activations get a
// bf16 image produced by the preceding kernel). Staging into LDS is pure
// 16-byte copies issued with global_load_async_to_lds_b128 (ASYNCcnt),
// double-buffered: one barrier per K-step, load/compute overlap.
// Block tile 64x128, wave tile 16x64 -> 4 WMMAs per K-step.
// ---------------------------------------------------------------------------

#define USE_ASYNC_LDS 1

static constexpr int Dn  = 512;     // model dim
static constexpr int Hh  = 8;       // heads
static constexpr int Nn  = 20000;   // nodes
static constexpr int Ee  = 160000;  // graph edges
static constexpr int ELl = 120000;  // local line-graph nodes
static constexpr int E2  = 200000;  // line-graph edges
static constexpr int MAXR = 120000; // max rows of any sublayer input

typedef __attribute__((ext_vector_type(16))) __bf16 v16bf;
typedef __attribute__((ext_vector_type(8)))  float  v8f;

union FragAB { v16bf v; uint4 q[2]; };

__device__ __forceinline__ unsigned short f2bf(float f) {
  unsigned u = __float_as_uint(f);
  u += 0x7fffu + ((u >> 16) & 1u);   // round-to-nearest-even
  return (unsigned short)(u >> 16);
}

// monotonic float<->uint encoding for atomic max on floats
__device__ __forceinline__ unsigned flipf(float f) {
  unsigned u = __float_as_uint(f);
  return (u & 0x80000000u) ? ~u : (u | 0x80000000u);
}
__device__ __forceinline__ float unflipf(unsigned u) {
  u = (u & 0x80000000u) ? (u & 0x7fffffffu) : ~u;
  return __uint_as_float(u);
}

// 16-byte global -> LDS copy. Async path uses the CDNA5 async-to-LDS DMA
// (tracked by ASYNCcnt); flat-LDS addresses truncate to the 32-bit LDS offset.
__device__ __forceinline__ void copy16(void* ldsDst, const void* gSrc) {
#if USE_ASYNC_LDS
  asm volatile("global_load_async_to_lds_b128 %0, %1, off"
               :: "v"((unsigned)(unsigned long long)ldsDst), "v"(gSrc)
               : "memory");
#else
  *(uint4*)ldsDst = *(const uint4*)gSrc;
#endif
}
__device__ __forceinline__ void asyncWait() {
#if USE_ASYNC_LDS
  asm volatile("s_wait_asynccnt 0x0" ::: "memory");
#endif
}

// ---------------------------------------------------------------------------
// Weight pre-pass: Wt[n*K + k] = bf16(W[k*N + n])  (transposed, bf16)
// ---------------------------------------------------------------------------
__global__ void cast_transpose_w(const float* __restrict__ W,
                                 unsigned short* __restrict__ Wt,
                                 int K, int N)
{
  long long i = (long long)blockIdx.x * blockDim.x + threadIdx.x;
  if (i >= (long long)K * N) return;
  const int k = (int)(i / N);
  const int n = (int)(i - (long long)k * N);
  Wt[(long long)n * K + k] = f2bf(W[i]);
}

// f32 -> bf16 elementwise, 8 elements per thread (n must be a multiple of 8)
__global__ void cast_bf16(const float* __restrict__ s,
                          unsigned short* __restrict__ d, long long n8)
{
  long long i = (long long)blockIdx.x * blockDim.x + threadIdx.x;
  if (i >= n8) return;
  const float4* sp = (const float4*)s + i * 2;
  float4 a = sp[0], b = sp[1];
  union { unsigned short u[8]; uint4 q; } r;
  r.u[0] = f2bf(a.x); r.u[1] = f2bf(a.y); r.u[2] = f2bf(a.z); r.u[3] = f2bf(a.w);
  r.u[4] = f2bf(b.x); r.u[5] = f2bf(b.y); r.u[6] = f2bf(b.z); r.u[7] = f2bf(b.w);
  *((uint4*)(d + i * 8)) = r.q;
}

// ---------------------------------------------------------------------------
// Tiled WMMA GEMM: C = A @ W (+bias) (+relu), bf16 inputs, f32 accumulate.
// A: bf16 [M][K]. Wt: bf16 [N][K]. C (f32) and/or Cb (bf16) outputs.
// Block: 256 threads (8 waves), 64x128 tile, K-step 32, double-buffered LDS.
// Wave w: rows (w&3)*16, cols (w>>2)*64 -> four 16x16 WMMA accumulators.
// Out-of-range A rows are clamped to M-1 (their C rows are never stored).
// ---------------------------------------------------------------------------
__global__ __launch_bounds__(256) void gemm_wmma_bf16(
    const unsigned short* __restrict__ A, const unsigned short* __restrict__ Wt,
    const float* __restrict__ bias, float* __restrict__ C,
    unsigned short* __restrict__ Cb,
    int M, int K, int N, int doRelu)
{
  __shared__ __align__(16) unsigned short lA[2][64 * 32];   // [row][k]
  __shared__ __align__(16) unsigned short lB[2][128 * 32];  // [col][k]

  const int t    = threadIdx.x;
  const int wave = t >> 5;
  const int lane = t & 31;
  const int bm = blockIdx.x * 64;
  const int bn = blockIdx.y * 128;
  const int wm = (wave & 3) * 16;
  const int wn = (wave >> 2) * 64;

  v8f acc[4] = {};

  // A fragment (16x32 bf16): lane l<16 -> row l, K {0..7,16..23};
  // lane l>=16 -> row l-16, K {8..15,24..31}.
  const int aRow  = wm + (lane & 15);
  const int aHalf = lane >> 4;
  const int aQ0 = (aRow * 64 + aHalf * 16) >> 4;        // uint4 index
  const int aQ1 = (aRow * 64 + 32 + aHalf * 16) >> 4;
  // B fragment (32x16 bf16): lane l -> col l&15, K half = l>>4
  const int bCol  = lane & 15;
  const int bHalf = lane >> 4;

  // staging: A tile 4KB -> 16B/thread; B tile 8KB -> 32B/thread
  const int saR = t >> 2;            // A row 0..63
  const int saC = (t & 3) * 8;       // A k-seg (8 bf16 = 16B)
  const int sgr = min(bm + saR, M - 1);
  const int sbC = t >> 1;            // B col 0..127
  const int sbK = (t & 1) * 16;      // B k-seg (16 bf16 = 32B)
  const unsigned short* aSrcBase = A + (long long)sgr * K + saC;
  const unsigned short* bSrcBase = Wt + (long long)(bn + sbC) * K + sbK;

  auto stage = [&](int buf, int k0) {
    copy16(&lA[buf][saR * 32 + saC], aSrcBase + k0);
    copy16(&lB[buf][sbC * 32 + sbK], bSrcBase + k0);
    copy16(&lB[buf][sbC * 32 + sbK + 8], bSrcBase + k0 + 8);
  };

  stage(0, 0);
  int buf = 0;
  for (int k0 = 0; k0 < K; k0 += 32, buf ^= 1) {
    asyncWait();
    __syncthreads();
    if (k0 + 32 < K) stage(buf ^ 1, k0 + 32);

    const uint4* lAq = (const uint4*)lA[buf];
    const uint4* lBq = (const uint4*)lB[buf];
    FragAB a;
    a.q[0] = lAq[aQ0];
    a.q[1] = lAq[aQ1];
    FragAB b[4];
#pragma unroll
    for (int tile = 0; tile < 4; ++tile) {
      const int qi = (wn + tile * 16 + bCol) * 4 + bHalf;
      b[tile].q[0] = lBq[qi];
      b[tile].q[1] = lBq[qi + 2];
    }
#pragma unroll
    for (int tile = 0; tile < 4; ++tile)
      acc[tile] = __builtin_amdgcn_wmma_f32_16x16x32_bf16(
          false, a.v, false, b[tile].v, (short)0, acc[tile], false, false);
  }

  // epilogue: lane<16 -> M=r, lane>=16 -> M=8+r; N = lane&15
  const int cn   = lane & 15;
  const int mOff = (lane >> 4) * 8;
#pragma unroll
  for (int tile = 0; tile < 4; ++tile) {
    const int gn = bn + wn + tile * 16 + cn;
    const float bv = bias ? bias[gn] : 0.0f;   // one load per tile
#pragma unroll
    for (int r = 0; r < 8; ++r) {
      const int gm = bm + wm + mOff + r;
      if (gm >= M) continue;
      float v = acc[tile][r] + bv;
      if (doRelu) v = fmaxf(v, 0.0f);
      if (C)  C[(long long)gm * N + gn] = v;
      if (Cb) Cb[(long long)gm * N + gn] = f2bf(v);
    }
  }
}

// ---------------------------------------------------------------------------
// fills
// ---------------------------------------------------------------------------
__global__ void fill_f32(float* __restrict__ p, float v, long long n) {
  long long i = (long long)blockIdx.x * blockDim.x + threadIdx.x;
  if (i < n) p[i] = v;
}
__global__ void fill_u32(unsigned* __restrict__ p, unsigned v, long long n) {
  long long i = (long long)blockIdx.x * blockDim.x + threadIdx.x;
  if (i < n) p[i] = v;
}

// ---------------------------------------------------------------------------
// attention: score[e,h] = <k[src]+rel, q[dst]>/8 ; atomic segment max over dst
// one wave per edge; lane covers 16 contiguous dims -> 4 lanes per head
// ---------------------------------------------------------------------------
__global__ __launch_bounds__(256) void attn_score(
    const float* __restrict__ q, const float* __restrict__ k,
    const float* __restrict__ relBase, const int* __restrict__ relIdx,
    const int* __restrict__ eSrc, const int* __restrict__ eDst,
    float* __restrict__ score, unsigned* __restrict__ smax, int nE)
{
  const int e = blockIdx.x * 8 + (threadIdx.x >> 5);
  if (e >= nE) return;
  const int lane = threadIdx.x & 31;
  const int s = eSrc[e], d = eDst[e];
  const long long rrow = relIdx ? (long long)relIdx[e] : (long long)e;
  const float4* kp = (const float4*)(k + (long long)s * Dn);
  const float4* qp = (const float4*)(q + (long long)d * Dn);
  const float4* rp = (const float4*)(relBase + rrow * Dn);
  float part = 0.0f;
  const int b4 = lane * 4;
#pragma unroll
  for (int i = 0; i < 4; ++i) {
    float4 kv = kp[b4 + i], rv = rp[b4 + i], qv = qp[b4 + i];
    part += (kv.x + rv.x) * qv.x + (kv.y + rv.y) * qv.y +
            (kv.z + rv.z) * qv.z + (kv.w + rv.w) * qv.w;
  }
  part += __shfl_xor(part, 1, 32);
  part += __shfl_xor(part, 2, 32);
  if ((lane & 3) == 0) {
    const int h = lane >> 2;
    const float sc = part * 0.125f;   // 1/sqrt(64)
    score[(long long)e * Hh + h] = sc;
    atomicMax(&smax[(long long)d * Hh + h], flipf(sc));
  }
}

// p = exp(score - smax[dst]); score <- p; z[dst] += p
__global__ void attn_expsum(float* __restrict__ score,
                            const int* __restrict__ eDst,
                            const unsigned* __restrict__ smax,
                            float* __restrict__ z, int nE)
{
  long long i = (long long)blockIdx.x * blockDim.x + threadIdx.x;
  if (i >= (long long)nE * Hh) return;
  const int e = (int)(i >> 3);
  const int h = (int)(i & 7);
  const int d = eDst[e];
  const float m = unflipf(smax[(long long)d * Hh + h]);
  const float p = __expf(score[i] - m);
  score[i] = p;
  atomicAdd(&z[(long long)d * Hh + h], p);
}

// out[dst] += alpha * (v[src]+rel); one wave per edge
__global__ __launch_bounds__(256) void attn_aggregate(
    const float* __restrict__ v, const float* __restrict__ relBase,
    const int* __restrict__ relIdx, const float* __restrict__ score,
    const float* __restrict__ z, const int* __restrict__ eSrc,
    const int* __restrict__ eDst, float* __restrict__ out, int nE)
{
  const int e = blockIdx.x * 8 + (threadIdx.x >> 5);
  if (e >= nE) return;
  const int lane = threadIdx.x & 31;
  const int s = eSrc[e], d = eDst[e];
  const long long rrow = relIdx ? (long long)relIdx[e] : (long long)e;
  const int h = lane >> 2;
  const float alpha = score[(long long)e * Hh + h] /
                      (z[(long long)d * Hh + h] + 1e-8f);
  const float4* vp = (const float4*)(v + (long long)s * Dn);
  const float4* rp = (const float4*)(relBase + rrow * Dn);
  float* op = out + (long long)d * Dn;
  const int b4 = lane * 4;
#pragma unroll
  for (int i = 0; i < 4; ++i) {
    float4 vv = vp[b4 + i], rv = rp[b4 + i];
    const int dd = (b4 + i) * 4;
    atomicAdd(&op[dd + 0], (vv.x + rv.x) * alpha);
    atomicAdd(&op[dd + 1], (vv.y + rv.y) * alpha);
    atomicAdd(&op[dd + 2], (vv.z + rv.z) * alpha);
    atomicAdd(&op[dd + 3], (vv.w + rv.w) * alpha);
  }
}

// ---------------------------------------------------------------------------
// out = LayerNorm(x + y) * g + b ; one wave per 512-float row.
// Optionally also writes a bf16 image (feeds the next GEMM's A operand).
// ---------------------------------------------------------------------------
__global__ __launch_bounds__(256) void ln_add(
    const float* __restrict__ x, const float* __restrict__ y,
    const float* __restrict__ g, const float* __restrict__ b,
    float* __restrict__ out, unsigned short* __restrict__ outb, int M)
{
  const int row = blockIdx.x * 8 + (threadIdx.x >> 5);
  if (row >= M) return;
  const int lane = threadIdx.x & 31;
  const float4* xp = (const float4*)(x + (long long)row * Dn);
  const float4* yp = (const float4*)(y + (long long)row * Dn);
  float4 v[4];
  float s = 0.0f;
#pragma unroll
  for (int i = 0; i < 4; ++i) {
    float4 a = xp[lane * 4 + i], c = yp[lane * 4 + i];
    v[i].x = a.x + c.x; v[i].y = a.y + c.y;
    v[i].z = a.z + c.z; v[i].w = a.w + c.w;
    s += v[i].x + v[i].y + v[i].z + v[i].w;
  }
#pragma unroll
  for (int m = 16; m >= 1; m >>= 1) s += __shfl_xor(s, m, 32);
  const float mean = s * (1.0f / Dn);
  float sq = 0.0f;
#pragma unroll
  for (int i = 0; i < 4; ++i) {
    float dx = v[i].x - mean, dy = v[i].y - mean;
    float dz = v[i].z - mean, dw = v[i].w - mean;
    sq += dx * dx + dy * dy + dz * dz + dw * dw;
  }
#pragma unroll
  for (int m = 16; m >= 1; m >>= 1) sq += __shfl_xor(sq, m, 32);
  const float inv = rsqrtf(sq * (1.0f / Dn) + 1e-5f);
  const float4* gp = (const float4*)g;
  const float4* bp = (const float4*)b;
  float4* op = (float4*)(out + (long long)row * Dn);
#pragma unroll
  for (int i = 0; i < 4; ++i) {
    float4 gv = gp[lane * 4 + i], bv = bp[lane * 4 + i], o;
    o.x = (v[i].x - mean) * inv * gv.x + bv.x;
    o.y = (v[i].y - mean) * inv * gv.y + bv.y;
    o.z = (v[i].z - mean) * inv * gv.z + bv.z;
    o.w = (v[i].w - mean) * inv * gv.w + bv.w;
    op[lane * 4 + i] = o;
    if (outb) {
      union { unsigned short u[4]; unsigned long long q; } r;
      r.u[0] = f2bf(o.x); r.u[1] = f2bf(o.y);
      r.u[2] = f2bf(o.z); r.u[3] = f2bf(o.w);
      *((unsigned long long*)(outb + (long long)row * Dn) + lane * 4 + i) = r.q;
    }
  }
}

// relIdx[e] = dst_ids[lg_src[e]]  (line-graph relation row in x)
__global__ void build_relidx(const int* __restrict__ lg_src,
                             const int* __restrict__ dst_ids,
                             int* __restrict__ relIdx, int n)
{
  int i = blockIdx.x * blockDim.x + threadIdx.x;
  if (i < n) relIdx[i] = dst_ids[lg_src[i]];
}

// out_lg[local_pos[r]] = local[r]   (float4 granularity)
__global__ void scatter_rows(float* __restrict__ outlg,
                             const float* __restrict__ local,
                             const int* __restrict__ pos, int rows)
{
  long long i = (long long)blockIdx.x * blockDim.x + threadIdx.x;
  if (i >= (long long)rows * (Dn / 4)) return;
  const int r  = (int)(i >> 7);
  const int c4 = (int)(i & 127);
  ((float4*)(outlg + (long long)pos[r] * Dn))[c4] =
      ((const float4*)(local + (long long)r * Dn))[c4];
}

// ---------------------------------------------------------------------------
// host side
// ---------------------------------------------------------------------------
struct Params {
  const float *Wq, *bq, *Wk, *Wv, *Wo, *bo, *ln1g, *ln1b,
              *W1, *b1, *W2, *b2, *ln2g, *ln2b;
};

static Params loadParams(void* const* d_in, int base) {
  Params p;
  p.Wq   = (const float*)d_in[base + 0];
  p.bq   = (const float*)d_in[base + 1];
  p.Wk   = (const float*)d_in[base + 2];
  p.Wv   = (const float*)d_in[base + 3];
  p.Wo   = (const float*)d_in[base + 4];
  p.bo   = (const float*)d_in[base + 5];
  p.ln1g = (const float*)d_in[base + 6];
  p.ln1b = (const float*)d_in[base + 7];
  p.W1   = (const float*)d_in[base + 8];
  p.b1   = (const float*)d_in[base + 9];
  p.W2   = (const float*)d_in[base + 10];
  p.b2   = (const float*)d_in[base + 11];
  p.ln2g = (const float*)d_in[base + 12];
  p.ln2b = (const float*)d_in[base + 13];
  return p;
}

static inline unsigned gblocks(long long n, int b) {
  return (unsigned)((n + b - 1) / b);
}

struct WtBufs {
  unsigned short *q, *k, *v, *o, *f1, *f2;  // bf16 transposed weights
};

static void run_sublayer(const float* X, int M,
                         const float* relBase, const int* relIdx,
                         const int* eSrc, const int* eDst, int nE,
                         const Params& p, float* outRow,
                         float* bufQ, float* bufK, float* bufV, float* bufO,
                         float* bufT, float* bufH,
                         unsigned short* abuf, unsigned short* fbuf,
                         float* score, unsigned* smax, float* zsum,
                         const WtBufs& wt, hipStream_t stream)
{
  const dim3 blk(256);

  // ---- weight pre-pass: bf16 transposed [N][K] ----
  cast_transpose_w<<<gblocks((long long)Dn * Dn, 256), blk, 0, stream>>>(
      p.Wq, wt.q, Dn, Dn);
  cast_transpose_w<<<gblocks((long long)Dn * Dn, 256), blk, 0, stream>>>(
      p.Wk, wt.k, Dn, Dn);
  cast_transpose_w<<<gblocks((long long)Dn * Dn, 256), blk, 0, stream>>>(
      p.Wv, wt.v, Dn, Dn);
  cast_transpose_w<<<gblocks((long long)Dn * Dn, 256), blk, 0, stream>>>(
      p.Wo, wt.o, Dn, Dn);
  cast_transpose_w<<<gblocks((long long)Dn * 2048, 256), blk, 0, stream>>>(
      p.W1, wt.f1, Dn, 2048);
  cast_transpose_w<<<gblocks((long long)2048 * Dn, 256), blk, 0, stream>>>(
      p.W2, wt.f2, 2048, Dn);

  // ---- init segment stats (flip(-inf) = 0x007FFFFF) + attention output ----
  fill_u32<<<gblocks((long long)M * Hh, 256), blk, 0, stream>>>(
      smax, 0x007FFFFFu, (long long)M * Hh);
  fill_f32<<<gblocks((long long)M * Hh, 256), blk, 0, stream>>>(
      zsum, 0.0f, (long long)M * Hh);
  fill_f32<<<gblocks((long long)M * Dn, 256), blk, 0, stream>>>(
      bufO, 0.0f, (long long)M * Dn);

  const dim3 gq((M + 63) / 64, Dn / 128);

  // ---- q,k,v projections (A = bf16 image of X) ----
  cast_bf16<<<gblocks((long long)M * Dn / 8, 256), blk, 0, stream>>>(
      X, abuf, (long long)M * Dn / 8);
  gemm_wmma_bf16<<<gq, blk, 0, stream>>>(abuf, wt.q, p.bq, bufQ, nullptr,
                                         M, Dn, Dn, 0);
  gemm_wmma_bf16<<<gq, blk, 0, stream>>>(abuf, wt.k, nullptr, bufK, nullptr,
                                         M, Dn, Dn, 0);
  gemm_wmma_bf16<<<gq, blk, 0, stream>>>(abuf, wt.v, nullptr, bufV, nullptr,
                                         M, Dn, Dn, 0);

  // ---- relation attention ----
  attn_score<<<gblocks(nE, 8), blk, 0, stream>>>(
      bufQ, bufK, relBase, relIdx, eSrc, eDst, score, smax, nE);
  attn_expsum<<<gblocks((long long)nE * Hh, 256), blk, 0, stream>>>(
      score, eDst, smax, zsum, nE);
  attn_aggregate<<<gblocks(nE, 8), blk, 0, stream>>>(
      bufV, relBase, relIdx, score, zsum, eSrc, eDst, bufO, nE);

  // ---- o @ Wo + residual LN ----
  cast_bf16<<<gblocks((long long)M * Dn / 8, 256), blk, 0, stream>>>(
      bufO, abuf, (long long)M * Dn / 8);
  gemm_wmma_bf16<<<gq, blk, 0, stream>>>(abuf, wt.o, p.bo, bufT, nullptr,
                                         M, Dn, Dn, 0);
  ln_add<<<gblocks(M, 8), blk, 0, stream>>>(X, bufT, p.ln1g, p.ln1b,
                                            bufH, abuf, M);

  // ---- FFN: relu(h@W1+b1)@W2+b2 + residual LN ----
  const dim3 gf((M + 63) / 64, 2048 / 128);
  gemm_wmma_bf16<<<gf, blk, 0, stream>>>(abuf, wt.f1, p.b1, nullptr, fbuf,
                                         M, Dn, 2048, 1);
  gemm_wmma_bf16<<<gq, blk, 0, stream>>>(fbuf, wt.f2, p.b2, bufT, nullptr,
                                         M, 2048, Dn, 0);
  ln_add<<<gblocks(M, 8), blk, 0, stream>>>(bufH, bufT, p.ln2g, p.ln2b,
                                            outRow, nullptr, M);
}

extern "C" void kernel_launch(void* const* d_in, const int* in_sizes, int n_in,
                              void* d_out, int out_size, void* d_ws, size_t ws_size,
                              hipStream_t stream) {
  const float* x          = (const float*)d_in[0];
  const float* lg_x       = (const float*)d_in[1];
  const float* lg_x_local = (const float*)d_in[2];
  const int* g_src     = (const int*)d_in[3];
  const int* g_dst     = (const int*)d_in[4];
  const int* lg_src    = (const int*)d_in[5];
  const int* lg_dst    = (const int*)d_in[6];
  const int* dst_ids   = (const int*)d_in[8];
  const int* local_pos = (const int*)d_in[9];
  const Params pn = loadParams(d_in, 10);
  const Params pe = loadParams(d_in, 24);

  // workspace carving (reused by both sublayers)
  char* w = (char*)d_ws;
  size_t off = 0;
  auto take = [&](size_t bytes) -> char* {
    char* p = w + off;
    off += (bytes + 255) & ~(size_t)255;
    return p;
  };
  float*    bufQ  = (float*)take((size_t)MAXR * Dn * 4);
  float*    bufK  = (float*)take((size_t)MAXR * Dn * 4);
  float*    bufV  = (float*)take((size_t)MAXR * Dn * 4);
  float*    bufO  = (float*)take((size_t)MAXR * Dn * 4);
  float*    bufT  = (float*)take((size_t)MAXR * Dn * 4);
  float*    bufH  = (float*)take((size_t)MAXR * Dn * 4);
  unsigned short* abuf = (unsigned short*)take((size_t)MAXR * Dn * 2);
  unsigned short* fbuf = (unsigned short*)take((size_t)MAXR * 2048 * 2);
  float*    score = (float*)take((size_t)E2 * Hh * 4);
  unsigned* smax  = (unsigned*)take((size_t)MAXR * Hh * 4);
  float*    zsum  = (float*)take((size_t)MAXR * Hh * 4);
  int*      relIdx = (int*)take((size_t)E2 * 4);

  WtBufs wt;
  wt.q  = (unsigned short*)take((size_t)Dn * Dn * 2);
  wt.k  = (unsigned short*)take((size_t)Dn * Dn * 2);
  wt.v  = (unsigned short*)take((size_t)Dn * Dn * 2);
  wt.o  = (unsigned short*)take((size_t)Dn * Dn * 2);
  wt.f1 = (unsigned short*)take((size_t)Dn * 2048 * 2);
  wt.f2 = (unsigned short*)take((size_t)2048 * Dn * 2);

  float* out_x     = (float*)d_out;
  float* out_lg    = out_x + (size_t)Nn * Dn;
  float* out_local = out_lg + (size_t)Ee * Dn;

  // ---- node sublayer: relations are lg_x rows (identity index) ----
  run_sublayer(x, Nn, lg_x, nullptr, g_src, g_dst, Ee, pn, out_x,
               bufQ, bufK, bufV, bufO, bufT, bufH, abuf, fbuf,
               score, smax, zsum, wt, stream);

  // ---- edge sublayer: relation row = x[dst_ids[lg_src[e]]] ----
  build_relidx<<<gblocks(E2, 256), 256, 0, stream>>>(lg_src, dst_ids, relIdx, E2);
  run_sublayer(lg_x_local, ELl, x, relIdx, lg_src, lg_dst, E2, pe, out_local,
               bufQ, bufK, bufV, bufO, bufT, bufH, abuf, fbuf,
               score, smax, zsum, wt, stream);

  // ---- out_lg_x = lg_x; out_lg_x[local_pos] = out_lg_x_local ----
  hipMemcpyAsync(out_lg, lg_x, (size_t)Ee * Dn * 4,
                 hipMemcpyDeviceToDevice, stream);
  scatter_rows<<<gblocks((long long)ELl * (Dn / 4), 256), 256, 0, stream>>>(
      out_lg, out_local, local_pos, ELl);
}
